// AtomAttentionEncoder_70574902608278
// MI455X (gfx1250) — matrix operand order repaired and checked by reading, not compiled, not run on hardware
//
#include <hip/hip_runtime.h>
#include <hip/hip_bf16.h>
#include <math.h>

typedef _Float16 half_t;
typedef _Float16 v16h __attribute__((ext_vector_type(16)));
typedef float    v8f  __attribute__((ext_vector_type(8)));

#define N_ATOM 2048
#define N_TOKEN 256
#define BLOCKN 24
#define C_ATOM 128
#define C_PAIR 32
#define C_TOKEN 384
#define NHEAD 4
#define DH 32

// ---------------------------------------------------------------------------
// Fragment index helpers (CDNA5 ISA 7.12.2, wave32)
// A 16-bit 16x32: lane m=lane&15, g=lane>>4; VGPR v holds K = kA(v,g), kA+1
__device__ __forceinline__ int kA(int v, int g) {
    return ((v < 4) ? 2 * v : 16 + 2 * (v - 4)) + 8 * g;
}
// B 16-bit 32x16: lane col=lane&15, g=lane>>4; VGPR v holds K = kA(v,g), kA+1
__device__ __forceinline__ int kB(int v, int g) { return kA(v, g); }

union PackH2 { half_t h[2]; unsigned int u; };
union PackH4 { half_t h[4]; uint2 u; };

// ---------------------------------------------------------------------------
// 1) atom embedding: c_atom0[n,c] = concat_feats(n) @ atom_W + atom_b + trunk_sing[tok[n]]
__global__ __launch_bounds__(128) void k_atom_embed(
    const float* __restrict__ pos, const float* __restrict__ charge,
    const float* __restrict__ elem, const float* __restrict__ name,
    const int* __restrict__ a2t, const float* __restrict__ tsing,
    const float* __restrict__ Wa, const float* __restrict__ ba,
    float* __restrict__ out) {
    const int n = blockIdx.x;
    const int c = threadIdx.x;
    __shared__ float xf[148];
    if (c < 3) xf[c] = pos[n * 3 + c];
    if (c == 3) xf[3] = charge[n];
    xf[4 + c] = elem[n * 128 + c];
    if (c < 16) xf[132 + c] = name[n * 16 + c];
    __syncthreads();
    float acc = ba[c];
#pragma unroll 4
    for (int k = 0; k < 148; ++k) acc = fmaf(xf[k], Wa[k * 128 + c], acc);
    const int t = a2t[n];
    out[(size_t)n * 128 + c] = acc + tsing[(size_t)t * 128 + c];
}

// ---------------------------------------------------------------------------
// 2) fused pair kernel: p_lm[i,j,:] = relu((pair_in @ pW + pb) @ W1 + b1) @ W2 + b2
//                                     + trunk_pair[tok_i, tok_j, :]
// One block per row i (4 waves). Each wave owns 16 consecutive j per tile and
// loops over 32 j-tiles. Layer0 (K=4) in VALU directly into the A fragment,
// then 4 WMMAs (32->64), relu via per-wave LDS round-trip, 4 WMMAs (64->32).
__global__ __launch_bounds__(128) void k_pair(
    const float* __restrict__ pos, const int* __restrict__ a2t,
    const float* __restrict__ tpair,
    const float* __restrict__ pW, const float* __restrict__ pb,
    const float* __restrict__ W1, const float* __restrict__ b1,
    const float* __restrict__ W2, const float* __restrict__ b2,
    float* __restrict__ plm) {
    const int i = blockIdx.x;
    const int wave = threadIdx.x >> 5;
    const int lane = threadIdx.x & 31;
    const int mloc = lane & 15;
    const int g = lane >> 4;

    __shared__ half_t hs[4][16 * 64];  // per-wave relu staging

    // hoisted per-lane layer0 weights for this lane's 16 K-channels
    float w0[16], w1r[16], w2r[16], w3r[16], pbr[16];
#pragma unroll
    for (int v = 0; v < 8; ++v) {
#pragma unroll
        for (int hh = 0; hh < 2; ++hh) {
            const int kk = kA(v, g) + hh;
            w0[2 * v + hh] = pW[0 * 32 + kk];
            w1r[2 * v + hh] = pW[1 * 32 + kk];
            w2r[2 * v + hh] = pW[2 * 32 + kk];
            w3r[2 * v + hh] = pW[3 * 32 + kk];
            pbr[2 * v + hh] = pb[kk];
        }
    }
    // B fragments: mlp1_W [32,64] -> 4 n-tiles; mlp2_W [64,32] -> 2 K x 2 N
    v16h bf1[4], bf2[2][2];
#pragma unroll
    for (int nt = 0; nt < 4; ++nt) {
#pragma unroll
        for (int v = 0; v < 8; ++v) {
            const int kk = kB(v, g);
            const int cc = nt * 16 + mloc;
            bf1[nt][2 * v] = (half_t)W1[kk * 64 + cc];
            bf1[nt][2 * v + 1] = (half_t)W1[(kk + 1) * 64 + cc];
        }
    }
#pragma unroll
    for (int kc = 0; kc < 2; ++kc)
#pragma unroll
        for (int nt = 0; nt < 2; ++nt)
#pragma unroll
            for (int v = 0; v < 8; ++v) {
                const int kk = kc * 32 + kB(v, g);
                const int cc = nt * 16 + mloc;
                bf2[kc][nt][2 * v] = (half_t)W2[kk * 32 + cc];
                bf2[kc][nt][2 * v + 1] = (half_t)W2[(kk + 1) * 32 + cc];
            }
    float rb1[4], rb2[2];
#pragma unroll
    for (int nt = 0; nt < 4; ++nt) rb1[nt] = b1[nt * 16 + mloc];
#pragma unroll
    for (int nt = 0; nt < 2; ++nt) rb2[nt] = b2[nt * 16 + mloc];

    const float pix = pos[i * 3 + 0], piy = pos[i * 3 + 1], piz = pos[i * 3 + 2];
    const int ti = a2t[i];

    for (int jt = 0; jt < 32; ++jt) {
        const int j0 = jt * 64 + wave * 16;
        // ---- layer0 into A fragment (this lane's row = pair (i, j0+mloc)) ----
        const int j = j0 + mloc;
        const float dx = pix - pos[j * 3 + 0];
        const float dy = piy - pos[j * 3 + 1];
        const float dz = piz - pos[j * 3 + 2];
        const float same = (ti == a2t[j]) ? 1.0f : 0.0f;
        v16h af;
#pragma unroll
        for (int e = 0; e < 16; ++e) {
            float pv = pbr[e];
            pv = fmaf(dx, w0[e], pv);
            pv = fmaf(dy, w1r[e], pv);
            pv = fmaf(dz, w2r[e], pv);
            pv = fmaf(same, w3r[e], pv);
            af[e] = (half_t)pv;
        }
        // ---- layer1: [16,32] @ [32,64] ----
        v8f h[4] = {v8f{}, v8f{}, v8f{}, v8f{}};
#pragma unroll
        for (int nt = 0; nt < 4; ++nt)
            h[nt] = __builtin_amdgcn_wmma_f32_16x16x32_f16(
                false, af, false, bf1[nt], (short)0, h[nt], false, false);
        // relu + bias, stage to per-wave LDS in row-major [16][64]
#pragma unroll
        for (int nt = 0; nt < 4; ++nt) {
            const int col = nt * 16 + mloc;
#pragma unroll
            for (int r = 0; r < 8; ++r) {
                float hv = h[nt][r] + rb1[nt];
                hv = fmaxf(hv, 0.0f);
                hs[wave][(r + 8 * g) * 64 + col] = (half_t)hv;
            }
        }
        // reload as two A fragments (K = 0..31 and 32..63)
        v16h af2[2];
#pragma unroll
        for (int kc = 0; kc < 2; ++kc)
#pragma unroll
            for (int v = 0; v < 8; ++v) {
                const int kk = kc * 32 + kA(v, g);
                af2[kc][2 * v] = hs[wave][mloc * 64 + kk];
                af2[kc][2 * v + 1] = hs[wave][mloc * 64 + kk + 1];
            }
        // ---- layer2: [16,64] @ [64,32] ----
        v8f o[2] = {v8f{}, v8f{}};
#pragma unroll
        for (int kc = 0; kc < 2; ++kc)
#pragma unroll
            for (int nt = 0; nt < 2; ++nt)
                o[nt] = __builtin_amdgcn_wmma_f32_16x16x32_f16(
                    false, af2[kc], false, bf2[kc][nt], (short)0, o[nt], false, false);
        // ---- epilogue: + b2 + trunk_pair gather, store ----
#pragma unroll
        for (int r = 0; r < 8; ++r) {
            const int jj = j0 + r + 8 * g;
            const int tj = a2t[jj];
            const size_t tbase = ((size_t)ti * N_TOKEN + tj) * 32;
            const size_t obase = ((size_t)i * N_ATOM + jj) * 32;
#pragma unroll
            for (int nt = 0; nt < 2; ++nt) {
                const int col = nt * 16 + mloc;
                plm[obase + col] = o[nt][r] + rb2[nt] + tpair[tbase + col];
            }
        }
    }
}

// ---------------------------------------------------------------------------
// 3) generic WMMA GEMM with f16 activations:
//    C[M,N] = act(A[M,K](f16) @ W[K,N](f32->f16) + bias [+ Res])
//    multiW: columns select among 3 weight/bias sets of width 128 (fused QKV)
//    Cf (f32) and/or Ch (f16) outputs, optional residual, optional relu.
// 128 threads (4 waves), 64x64 tile, K-step 32.
__global__ __launch_bounds__(128) void k_gemm(
    const half_t* __restrict__ A,
    const float* __restrict__ W0, const float* __restrict__ W1p,
    const float* __restrict__ W2p,
    const float* __restrict__ b0, const float* __restrict__ b1p,
    const float* __restrict__ b2p,
    const float* __restrict__ Res, float* __restrict__ Cf,
    half_t* __restrict__ Ch,
    int M, int N, int K, int relu, int multiW) {
    __shared__ half_t As[64 * 32];
    __shared__ half_t Ws[32 * 64];
    const int tid = threadIdx.x;
    const int wave = tid >> 5;
    const int lane = tid & 31;
    const int mloc = lane & 15;
    const int g = lane >> 4;
    const int m0 = blockIdx.y * 64;
    const int n0 = blockIdx.x * 64;

    v8f acc[4] = {v8f{}, v8f{}, v8f{}, v8f{}};
    for (int k0 = 0; k0 < K; k0 += 32) {
        // stage A: 64x32 f16 as 1024 dword copies (2 halfs each)
#pragma unroll
        for (int t = tid; t < 1024; t += 128) {
            const int r = t >> 4, c = (t & 15) * 2;
            *(unsigned int*)&As[r * 32 + c] =
                *(const unsigned int*)(A + (size_t)(m0 + r) * K + (k0 + c));
        }
        // stage W: 32x64, f32 -> packed f16
#pragma unroll
        for (int t = tid; t < 1024; t += 128) {
            const int r = t >> 5, c = (t & 31) * 2;
            int cc = n0 + c, sel = 0;
            if (multiW) { sel = cc >> 7; cc &= 127; }
            const float* Wp = (sel == 0) ? W0 : ((sel == 1) ? W1p : W2p);
            const int ldw = multiW ? 128 : N;
            PackH2 p;
            p.h[0] = (half_t)Wp[(size_t)(k0 + r) * ldw + cc];
            p.h[1] = (half_t)Wp[(size_t)(k0 + r) * ldw + cc + 1];
            *(unsigned int*)&Ws[r * 64 + c] = p.u;
        }
        // prefetch next K tile of A (global_prefetch_b8)
        if (k0 + 32 < K)
            __builtin_prefetch(A + (size_t)(m0 + (tid >> 1)) * K + (k0 + 32), 0, 0);
        __syncthreads();
        v16h af;
#pragma unroll
        for (int v = 0; v < 8; ++v) {
            const int kk = kA(v, g);
            af[2 * v] = As[(wave * 16 + mloc) * 32 + kk];
            af[2 * v + 1] = As[(wave * 16 + mloc) * 32 + kk + 1];
        }
#pragma unroll
        for (int nt = 0; nt < 4; ++nt) {
            v16h bf;
#pragma unroll
            for (int v = 0; v < 8; ++v) {
                const int kk = kB(v, g);
                const int cc = nt * 16 + mloc;
                bf[2 * v] = Ws[kk * 64 + cc];
                bf[2 * v + 1] = Ws[(kk + 1) * 64 + cc];
            }
            acc[nt] = __builtin_amdgcn_wmma_f32_16x16x32_f16(
                false, af, false, bf, (short)0, acc[nt], false, false);
        }
        __syncthreads();
    }
#pragma unroll
    for (int nt = 0; nt < 4; ++nt) {
        const int col = n0 + nt * 16 + mloc;
        int ccl = col, sel = 0;
        if (multiW) { sel = col >> 7; ccl = col & 127; }
        const float* bp = (sel == 0) ? b0 : ((sel == 1) ? b1p : b2p);
        const float bv = bp ? bp[ccl] : 0.0f;
#pragma unroll
        for (int r = 0; r < 8; ++r) {
            const int row = m0 + wave * 16 + r + 8 * g;
            float val = acc[nt][r] + bv;
            if (Res) val += Res[(size_t)row * N + col];
            if (relu) val = fmaxf(val, 0.0f);
            if (Cf) Cf[(size_t)row * N + col] = val;
            if (Ch) Ch[(size_t)row * N + col] = (half_t)val;
        }
    }
}

// ---------------------------------------------------------------------------
// 4) LayerNorm over C_ATOM=128 -> f16 output: one wave per row, 4 ch/lane
__global__ __launch_bounds__(256) void k_ln(
    const float* __restrict__ X, const float* __restrict__ S,
    const float* __restrict__ B, half_t* __restrict__ Y) {
    const int row = blockIdx.x * 8 + (threadIdx.x >> 5);
    const int lane = threadIdx.x & 31;
    const float4 xv = ((const float4*)(X + (size_t)row * 128))[lane];
    float s = xv.x + xv.y + xv.z + xv.w;
    float s2 = xv.x * xv.x + xv.y * xv.y + xv.z * xv.z + xv.w * xv.w;
#pragma unroll
    for (int off = 16; off; off >>= 1) {
        s += __shfl_xor(s, off);
        s2 += __shfl_xor(s2, off);
    }
    const float mean = s * (1.0f / 128.0f);
    const float var = s2 * (1.0f / 128.0f) - mean * mean;
    const float inv = rsqrtf(var + 1e-5f);
    const float4 sv = ((const float4*)S)[lane];
    const float4 bv = ((const float4*)B)[lane];
    PackH4 p;
    p.h[0] = (half_t)((xv.x - mean) * inv * sv.x + bv.x);
    p.h[1] = (half_t)((xv.y - mean) * inv * sv.y + bv.y);
    p.h[2] = (half_t)((xv.z - mean) * inv * sv.z + bv.z);
    p.h[3] = (half_t)((xv.w - mean) * inv * sv.w + bv.w);
    ((uint2*)(Y + (size_t)row * 128))[lane] = p.u;
}

// ---------------------------------------------------------------------------
// 5) local block attention over interleaved qkv[n, 3*128]; f16 output.
// One wave per (atom, head): lanes 0..23 = neighbors for scores/softmax;
// lanes 0..31 = head-dim for the weighted sum.
__global__ __launch_bounds__(256) void k_attn(
    const float* __restrict__ qkv, const float* __restrict__ plm,
    const int* __restrict__ bidx, const float* __restrict__ Wb,
    const float* __restrict__ bb, half_t* __restrict__ ob) {
    const int wid = blockIdx.x * 8 + (threadIdx.x >> 5);
    const int n = wid >> 2;
    const int h = wid & 3;
    const int lane = threadIdx.x & 31;

    float score = -INFINITY;
    int nb = 0;
    if (lane < BLOCKN) {
        nb = bidx[n * BLOCKN + lane];
        const float* qp = qkv + (size_t)n * 384 + h * DH;
        const float* kp = qkv + (size_t)nb * 384 + 128 + h * DH;
        float s = 0.0f;
#pragma unroll 8
        for (int d = 0; d < DH; ++d) s = fmaf(qp[d], kp[d], s);
        const float* pp = plm + ((size_t)n * N_ATOM + nb) * C_PAIR;
        float bsum = bb[h];
#pragma unroll 8
        for (int c = 0; c < C_PAIR; ++c) bsum = fmaf(pp[c], Wb[c * NHEAD + h], bsum);
        score = s * 0.17677669529663687f + bsum;  // 1/sqrt(32)
    }
    float mx = score;
#pragma unroll
    for (int off = 16; off; off >>= 1) mx = fmaxf(mx, __shfl_xor(mx, off));
    const float e = (lane < BLOCKN) ? __expf(score - mx) : 0.0f;
    float ssum = e;
#pragma unroll
    for (int off = 16; off; off >>= 1) ssum += __shfl_xor(ssum, off);
    const float attn = e / ssum;

    float o = 0.0f;
    for (int b = 0; b < BLOCKN; ++b) {
        const float a = __shfl(attn, b);
        const int nbb = __shfl(nb, b);
        o = fmaf(a, qkv[(size_t)nbb * 384 + 256 + h * DH + lane], o);
    }
    ob[(size_t)n * C_ATOM + h * DH + lane] = (half_t)o;
}

// ---------------------------------------------------------------------------
// 6) segment mean over sorted atom_to_token: one block per token
__global__ __launch_bounds__(128) void k_seg_mean(
    const float* __restrict__ proj, const int* __restrict__ a2t,
    float* __restrict__ out) {
    const int t = blockIdx.x;
    int lo = 0, hi = N_ATOM;
    while (lo < hi) { const int mid = (lo + hi) >> 1; if (a2t[mid] < t) lo = mid + 1; else hi = mid; }
    const int s = lo;
    lo = s; hi = N_ATOM;
    while (lo < hi) { const int mid = (lo + hi) >> 1; if (a2t[mid] < t + 1) lo = mid + 1; else hi = mid; }
    const int e = lo;
    const float invc = 1.0f / fmaxf((float)(e - s), 1.0f);
    for (int c = threadIdx.x; c < C_TOKEN; c += 128) {
        float acc = 0.0f;
        for (int a = s; a < e; ++a) acc += proj[(size_t)a * C_TOKEN + c];
        out[(size_t)t * C_TOKEN + c] = acc * invc;
    }
}

// ---------------------------------------------------------------------------
extern "C" void kernel_launch(void* const* d_in, const int* in_sizes, int n_in,
                              void* d_out, int out_size, void* d_ws, size_t ws_size,
                              hipStream_t stream) {
    // inputs (setup_inputs dict order, params flattened in insertion order)
    const float* ref_pos = (const float*)d_in[0];
    const float* ref_charge = (const float*)d_in[1];
    const float* ref_element = (const float*)d_in[2];
    const float* ref_name = (const float*)d_in[3];
    const int* a2t = (const int*)d_in[4];
    // d_in[5] = restype (only shape used)
    const float* trunk_sing = (const float*)d_in[6];
    const float* trunk_pair = (const float*)d_in[7];
    const int* block_index = (const int*)d_in[8];
    const float* atom_W = (const float*)d_in[9];
    const float* atom_b = (const float*)d_in[10];
    const float* pair_W = (const float*)d_in[11];
    const float* pair_b = (const float*)d_in[12];
    const float* mlp1_W = (const float*)d_in[13];
    const float* mlp1_b = (const float*)d_in[14];
    const float* mlp2_W = (const float*)d_in[15];
    const float* mlp2_b = (const float*)d_in[16];
    const float* post_W = (const float*)d_in[17];
    const float* post_b = (const float*)d_in[18];
    const float* ln1_s = (const float*)d_in[19];
    const float* ln1_b = (const float*)d_in[20];
    const float* Wq = (const float*)d_in[21];
    const float* bq = (const float*)d_in[22];
    const float* Wk = (const float*)d_in[23];
    const float* bk = (const float*)d_in[24];
    const float* Wv = (const float*)d_in[25];
    const float* bv = (const float*)d_in[26];
    const float* Wb = (const float*)d_in[27];
    const float* bb = (const float*)d_in[28];
    const float* Wo = (const float*)d_in[29];
    const float* bo = (const float*)d_in[30];
    const float* ln2_s = (const float*)d_in[31];
    const float* ln2_b = (const float*)d_in[32];
    const float* W1 = (const float*)d_in[33];
    const float* b1 = (const float*)d_in[34];
    const float* W2 = (const float*)d_in[35];
    const float* b2 = (const float*)d_in[36];

    // outputs concatenated: a_token, q_atom, c_atom0, p_lm
    float* out = (float*)d_out;
    float* a_token = out;                        // 256*384
    float* q_atom = out + 98304;                 // 2048*128 (residual stream)
    float* c_atom0 = out + 98304 + 262144;       // 2048*128
    float* plm = out + 98304 + 262144 + 262144;  // 2048*2048*32

    // workspace layout (bytes)
    char* wsb = (char*)d_ws;
    float* qkv = (float*)wsb;                    // 2048*384 f32  (3,145,728 B)
    half_t* h16 = (half_t*)(wsb + 3145728);      // 2048*128 f16  (524,288 B)
    half_t* o16 = (half_t*)(wsb + 3670016);      // 2048*128 f16
    half_t* t16 = (half_t*)(wsb + 4194304);      // 2048*512 f16  (2,097,152 B)
    half_t* x16 = (half_t*)(wsb + 6291456);      // 2048*128 f16
    float* proj = (float*)(wsb + 6815744);       // 2048*384 f32

    // 1) atom embedding (+trunk_sing)
    k_atom_embed<<<N_ATOM, 128, 0, stream>>>(ref_pos, ref_charge, ref_element,
                                             ref_name, a2t, trunk_sing, atom_W,
                                             atom_b, c_atom0);
    // 2) fused pair pipeline (+trunk_pair gather) -> p_lm
    k_pair<<<N_ATOM, 128, 0, stream>>>(ref_pos, a2t, trunk_pair, pair_W, pair_b,
                                       mlp1_W, mlp1_b, mlp2_W, mlp2_b, plm);
    // x = c_atom0 (residual stream lives in the q_atom output region)
    hipMemcpyAsync(q_atom, c_atom0, (size_t)262144 * sizeof(float),
                   hipMemcpyDeviceToDevice, stream);

    // 3) atom transformer, 3 layers
    for (int l = 0; l < 3; ++l) {
        k_ln<<<256, 256, 0, stream>>>(q_atom, ln1_s + l * 128, ln1_b + l * 128, h16);
        // fused QKV projection -> interleaved qkv[n, 3*128]
        k_gemm<<<dim3(6, 32), 128, 0, stream>>>(
            h16, Wq + (size_t)l * 16384, Wk + (size_t)l * 16384,
            Wv + (size_t)l * 16384, bq + l * 128, bk + l * 128, bv + l * 128,
            nullptr, qkv, nullptr, N_ATOM, 384, 128, 0, 1);
        k_attn<<<1024, 256, 0, stream>>>(qkv, plm, block_index, Wb + l * 128,
                                         bb + l * 4, o16);
        // x += o @ Wo + bo
        k_gemm<<<dim3(2, 32), 128, 0, stream>>>(
            o16, Wo + (size_t)l * 16384, nullptr, nullptr, bo + l * 128, nullptr,
            nullptr, q_atom, q_atom, nullptr, N_ATOM, 128, 128, 0, 0);
        k_ln<<<256, 256, 0, stream>>>(q_atom, ln2_s + l * 128, ln2_b + l * 128, h16);
        // t = relu(h2 @ W1 + b1), kept f16
        k_gemm<<<dim3(8, 32), 128, 0, stream>>>(
            h16, W1 + (size_t)l * 65536, nullptr, nullptr, b1 + l * 512, nullptr,
            nullptr, nullptr, nullptr, t16, N_ATOM, 512, 128, 1, 0);
        // x += t @ W2 + b2 ; last layer also emits f16 copy of x for post GEMM
        k_gemm<<<dim3(2, 32), 128, 0, stream>>>(
            t16, W2 + (size_t)l * 65536, nullptr, nullptr, b2 + l * 128, nullptr,
            nullptr, q_atom, q_atom, (l == 2) ? x16 : nullptr,
            N_ATOM, 128, 512, 0, 0);
    }

    // 4) post projection + segment mean -> a_token
    k_gemm<<<dim3(6, 32), 128, 0, stream>>>(
        x16, post_W, nullptr, nullptr, post_b, nullptr, nullptr,
        nullptr, proj, nullptr, N_ATOM, C_TOKEN, 128, 0, 0);
    k_seg_mean<<<N_TOKEN, 128, 0, stream>>>(proj, a2t, a_token);
}
